// Attention_10144712753719
// MI455X (gfx1250) — matrix operand order repaired and checked by reading, not compiled, not run on hardware
//
#include <hip/hip_runtime.h>
#include <hip/hip_bf16.h>
#include <stdint.h>

// ---------------------------------------------------------------------------
// Problem constants (reference: B=8, T=2048, E=2048, D=256; T==E required)
// ---------------------------------------------------------------------------
static constexpr int B_ = 8;
static constexpr int T_ = 2048;
static constexpr int E_ = 2048;
static constexpr int D_ = 256;
static constexpr int M_ = B_ * T_;   // 16384 total rows

typedef __bf16          bf16_t;
typedef bf16_t          v16bf  __attribute__((ext_vector_type(16)));
typedef float           v8f    __attribute__((ext_vector_type(8)));
typedef unsigned short  u16x16 __attribute__((ext_vector_type(16)));
typedef unsigned short  u16x8  __attribute__((ext_vector_type(8)));

__device__ __forceinline__ unsigned short f2bf(float f) {
  uint32_t u = __builtin_bit_cast(uint32_t, f);
  u += 0x7fffu + ((u >> 16) & 1u);            // round-to-nearest-even
  return (unsigned short)(u >> 16);
}

__device__ __forceinline__ v8f wmma_bf16(u16x16 a, u16x16 b, v8f c) {
  return __builtin_amdgcn_wmma_f32_16x16x32_bf16(
      false, __builtin_bit_cast(v16bf, a),
      false, __builtin_bit_cast(v16bf, b),
      (short)0, c, false, false);
}

// ---------------------------------------------------------------------------
// gfx1250 async global->LDS copy (ASYNCcnt-tracked). One b128 per lane.
// ---------------------------------------------------------------------------
__device__ __forceinline__ void async_load_b128(uint32_t lds_off, const void* gptr) {
  asm volatile("global_load_async_to_lds_b128 %0, %1, off"
               :: "v"(lds_off), "v"(gptr)
               : "memory");
}
__device__ __forceinline__ void wait_asynccnt0() {
  asm volatile("s_wait_asynccnt 0" ::: "memory");
}
__device__ __forceinline__ void wait_dscnt0() {
  asm volatile("s_wait_dscnt 0" ::: "memory");
}

// A-operand (16x32 bf16): rows m0..m0+15, K-chunk k0..k0+31 from row-major [.,ld].
// Layout: lane<16 -> M=l, K0..15 contiguous ; lane>=16 -> K16..31
__device__ __forceinline__ u16x16 load_a(const unsigned short* __restrict__ base,
                                         int ld, int m0, int k0, int l, int half) {
  const unsigned short* p = base + (size_t)(m0 + l) * ld + k0 + half * 16;
  u16x8 lo = *(const u16x8*)p;
  u16x8 hi = *(const u16x8*)(p + 8);
  u16x16 r;
#pragma unroll
  for (int i = 0; i < 8; ++i) { r[i] = lo[i]; r[8 + i] = hi[i]; }
  return r;
}

// B-operand from 16 contiguous bf16 (lane<16 -> N=l, K k0..k0+15; lane>=16 -> K16..31).
// p must already point at this lane's 16 elements (row n0+l, offset k0 + half*16).
__device__ __forceinline__ u16x16 load_b16(const unsigned short* p) {
  u16x8 lo = *(const u16x8*)p;
  u16x8 hi = *(const u16x8*)(p + 8);
  u16x16 r;
#pragma unroll
  for (int i = 0; i < 8; ++i) { r[i] = lo[i]; r[8 + i] = hi[i]; }
  return r;
}

// ---------------------------------------------------------------------------
// Kernel 1: f32 -> bf16 copy (x)
// ---------------------------------------------------------------------------
__global__ void cvt_f32_bf16(const float* __restrict__ in,
                             unsigned short* __restrict__ out, int n) {
  int i = blockIdx.x * blockDim.x + threadIdx.x;
  if (i < n) out[i] = f2bf(in[i]);
}

// Kernel 2: W [E,D] f32 -> Wt [D,E] bf16 (transposed so B-operand reads are contiguous)
__global__ void cvt_w_t(const float* __restrict__ w,
                        unsigned short* __restrict__ wt) {
  int i = blockIdx.x * blockDim.x + threadIdx.x;   // over E_*D_
  int e = i >> 8;           // / D_
  int d = i & (D_ - 1);
  wt[(size_t)d * E_ + e] = f2bf(w[i]);
}

// ---------------------------------------------------------------------------
// Kernel 3: projections  C = Xbf @ W  via WMMA bf16.
// grid = (M/128, D/64, 3); block = 256 (8 waves). z: 0->K, 1->Q, 2->V(transposed)
// W tile (64 cols x 128 K) staged cooperatively in LDS via async DMA, shared by
// all 8 waves. B operand software-pipelined (next ds_load pair in flight while
// current WMMA executes).
// ---------------------------------------------------------------------------
__global__ __launch_bounds__(256) void proj_kernel(
    const unsigned short* __restrict__ xb,
    const unsigned short* __restrict__ wtk,
    const unsigned short* __restrict__ wtq,
    const unsigned short* __restrict__ wtv,
    unsigned short* __restrict__ kb,
    unsigned short* __restrict__ qb,
    unsigned short* __restrict__ vt) {
  __shared__ alignas(16) unsigned short lds_w[64 * 128];   // 16 KB

  const int tid  = threadIdx.x;
  const int wave = tid >> 5;
  const int lane = tid & 31;
  const int l = lane & 15, half = lane >> 4;
  const int m0 = blockIdx.x * 128 + wave * 16;
  const int n0 = blockIdx.y * 64;
  const int z  = blockIdx.z;
  const unsigned short* wt = (z == 0) ? wtk : (z == 1) ? wtq : wtv;
  const uint32_t ldsw0 = (uint32_t)(uintptr_t)(void*)lds_w;

  // cooperative async stage of W rows n0..n0+63, K-chunk k0..k0+127 (16 KB)
  auto stage_w = [&](int k0) {
#pragma unroll
    for (int i = 0; i < 4; ++i) {
      const int c = i * 256 + tid;        // 1024 x b128 chunks
      const int row = c >> 4, col = c & 15;
      async_load_b128(ldsw0 + row * 256 + col * 16,
                      (const char*)wt + ((size_t)(n0 + row) * E_ + k0) * 2 + col * 16);
    }
  };

  stage_w(0);

  v8f acc[4];
  v8f zero = {};
#pragma unroll
  for (int i = 0; i < 4; ++i) acc[i] = zero;

  for (int it = 0; it < 16; ++it) {       // K loop, 128 per step
    const int k0 = it * 128;
    if (it + 1 < 16)                      // global_prefetch_b8 next A rows
      __builtin_prefetch((const char*)xb + ((size_t)(m0 + l) * E_ + k0 + 128) * 2, 0, 1);

    wait_asynccnt0();
    __syncthreads();                      // W stage visible to all waves

    u16x16 a[4];
#pragma unroll
    for (int cc = 0; cc < 4; ++cc)        // batch-issue all A loads up front
      a[cc] = load_a(xb, E_, m0, k0 + cc * 32, l, half);

    // software-pipelined B operand: 16 WMMAs (cc x nt), next tile always in flight
    u16x16 bc = load_b16(lds_w + (l) * 128 + half * 16);   // cc=0, nt=0
#pragma unroll
    for (int idx = 0; idx < 16; ++idx) {
      const int cc = idx >> 2, nt = idx & 3;
      u16x16 bn = bc;
      if (idx < 15) {
        const int cc2 = (idx + 1) >> 2, nt2 = (idx + 1) & 3;
        bn = load_b16(lds_w + (nt2 * 16 + l) * 128 + cc2 * 32 + half * 16);
      }
      acc[nt] = wmma_bf16(a[cc], bc, acc[nt]);
      bc = bn;
    }

    __syncthreads();                      // all waves done reading lds_w
    if (it + 1 < 16) stage_w(k0 + 128);   // overlap next DMA with loop overhead
  }

#pragma unroll
  for (int nt = 0; nt < 4; ++nt) {
#pragma unroll
    for (int r = 0; r < 8; ++r) {
      const int m = m0 + r + 8 * half;          // C-layout: VGPR r -> M = r + 8*half
      const int n = n0 + nt * 16 + l;           //            N = lane&15
      const unsigned short h = f2bf(acc[nt][r]);
      if (z == 0)      kb[(size_t)m * D_ + n] = h;
      else if (z == 1) qb[(size_t)m * D_ + n] = h;
      else {
        const int b = m >> 11;                  // / T_
        const int t = m & (T_ - 1);
        vt[((size_t)b * D_ + n) * T_ + t] = h;  // V stored [B,D,T] (transposed)
      }
    }
  }
}

// ---------------------------------------------------------------------------
// Kernel 4: flash attention. scores = K·Qᵀ/sqrt(D), causal mask, softmax, @V.
// grid = (T/128, B); block = 256 (8 waves, each owns 16 t-rows, full D=256).
// Q block (32x256) and V block (256x32) staged per s-block via async DMA and
// shared by all 8 waves. Workgroup-uniform trip count (fully-masked blocks are
// absorbed by the online softmax: alpha=1, p=0). B operands software-pipelined.
// ---------------------------------------------------------------------------
__global__ __launch_bounds__(256) void fa_kernel(
    const unsigned short* __restrict__ kb,
    const unsigned short* __restrict__ qb,
    const unsigned short* __restrict__ vt,
    float* __restrict__ out) {
  __shared__ alignas(16) unsigned short lds_q[32 * 256];   // 16 KB
  __shared__ alignas(16) unsigned short lds_v[256 * 32];   // 16 KB
  __shared__ alignas(16) unsigned short lds_p[8 * 16 * 32];//  8 KB (per-wave P)

  const int tid  = threadIdx.x;
  const int wave = tid >> 5;
  const int lane = tid & 31;
  const int l = lane & 15, half = lane >> 4;
  const int b  = blockIdx.y;
  const int t0 = blockIdx.x * 128 + wave * 16;

  const unsigned short* kbase = kb + (size_t)b * T_ * D_;
  const unsigned short* qbase = qb + (size_t)b * T_ * D_;
  const unsigned short* vbase = vt + (size_t)b * D_ * T_;
  const uint32_t ldsq0 = (uint32_t)(uintptr_t)(void*)lds_q;
  const uint32_t ldsv0 = (uint32_t)(uintptr_t)(void*)lds_v;

  // cooperative async stage of q rows s0..s0+31 (full D) and vt rows 0..255 (32 s)
  auto stage_qv = [&](int s0) {
#pragma unroll
    for (int i = 0; i < 4; ++i) {
      const int c = i * 256 + tid;
      { const int row = c >> 5, col = c & 31;     // q: 32 rows x 512 B
        async_load_b128(ldsq0 + row * 512 + col * 16,
                        (const char*)qbase + ((size_t)(s0 + row) * D_) * 2 + col * 16); }
      { const int row = c >> 2, col = c & 3;      // v: 256 rows x 64 B
        async_load_b128(ldsv0 + row * 64 + col * 16,
                        (const char*)vbase + ((size_t)row * T_ + s0) * 2 + col * 16); }
    }
  };

  // Preload this wave's 16 K-rows (full D=256) as 8 resident A-operand chunks.
  u16x16 ka[8];
#pragma unroll
  for (int c = 0; c < 8; ++c) ka[c] = load_a(kbase, D_, t0, c * 32, l, half);

  v8f o[16];
  v8f zero = {};
#pragma unroll
  for (int i = 0; i < 16; ++i) o[i] = zero;
  float mrow[8], lrow[8];
#pragma unroll
  for (int r = 0; r < 8; ++r) { mrow[r] = -INFINITY; lrow[r] = 0.0f; }

  unsigned short* pl = lds_p + wave * (16 * 32);
  const float scale = 0.0625f;                    // 1/sqrt(256)

  // workgroup-uniform trip count: covers s <= max row of the block
  const int nblk = ((blockIdx.x * 128 + 112) >> 5) + 1;
  stage_qv(0);

  for (int sb = 0; sb < nblk; ++sb) {
    const int s0 = sb * 32;

    wait_asynccnt0();
    __syncthreads();                              // q/v stage visible

    // S = K · Qᵀ (two 16x16 N-tiles, K-depth 256); B pipelined from LDS
    v8f st0 = zero, st1 = zero;
    u16x16 bq = load_b16(lds_q + l * 256 + half * 16);     // c=0, j=0
#pragma unroll
    for (int idx = 0; idx < 16; ++idx) {
      const int c = idx >> 1, j = idx & 1;
      u16x16 bn = bq;
      if (idx < 15) {
        const int c2 = (idx + 1) >> 1, j2 = (idx + 1) & 1;
        bn = load_b16(lds_q + (j2 * 16 + l) * 256 + c2 * 32 + half * 16);
      }
      if (j == 0) st0 = wmma_bf16(ka[c], bq, st0);
      else        st1 = wmma_bf16(ka[c], bq, st1);
      bq = bn;
    }

    float s0v[8], s1v[8];
#pragma unroll
    for (int r = 0; r < 8; ++r) { s0v[r] = st0[r] * scale; s1v[r] = st1[r] * scale; }

    if (s0 + 31 > t0) {                           // diagonal / fully-masked blocks
#pragma unroll
      for (int r = 0; r < 8; ++r) {
        const int trow = t0 + r + 8 * half;
        if (s0 + l      > trow) s0v[r] = -INFINITY;
        if (s0 + 16 + l > trow) s1v[r] = -INFINITY;
      }
    }

    // Online softmax: row stats replicated across each 16-lane half.
#pragma unroll
    for (int r = 0; r < 8; ++r) {
      float bm = fmaxf(s0v[r], s1v[r]);
      bm = fmaxf(bm, __shfl_xor(bm, 1, 32));
      bm = fmaxf(bm, __shfl_xor(bm, 2, 32));
      bm = fmaxf(bm, __shfl_xor(bm, 4, 32));
      bm = fmaxf(bm, __shfl_xor(bm, 8, 32));
      const float mnew  = fmaxf(mrow[r], bm);
      const float alpha = __expf(mrow[r] - mnew);
      mrow[r] = mnew;
      const float p0 = __expf(s0v[r] - mnew);
      const float p1 = __expf(s1v[r] - mnew);
      float rs = p0 + p1;
      rs += __shfl_xor(rs, 1, 32);
      rs += __shfl_xor(rs, 2, 32);
      rs += __shfl_xor(rs, 4, 32);
      rs += __shfl_xor(rs, 8, 32);
      lrow[r] = lrow[r] * alpha + rs;
#pragma unroll
      for (int dt = 0; dt < 16; ++dt) o[dt][r] *= alpha;    // rescale O row
      const int m = r + 8 * half;                            // stage P (bf16) to LDS
      pl[m * 32 + l]      = f2bf(p0);
      pl[m * 32 + 16 + l] = f2bf(p1);
    }

    wait_dscnt0();                                // P visible across lanes (same wave)

    // Reload P in A-operand layout straight from LDS (ds_load_b128 x2).
    const unsigned short* pp = pl + l * 32 + half * 16;
    u16x8 plo = *(const u16x8*)pp;
    u16x8 phi = *(const u16x8*)(pp + 8);
    u16x16 pa;
#pragma unroll
    for (int i = 0; i < 8; ++i) { pa[i] = plo[i]; pa[8 + i] = phi[i]; }

    // O += P · V (16 N-tiles over D=256); B pipelined from LDS
    u16x16 bv = load_b16(lds_v + l * 32 + half * 16);      // dt=0
#pragma unroll
    for (int dt = 0; dt < 16; ++dt) {
      u16x16 bn = bv;
      if (dt < 15)
        bn = load_b16(lds_v + ((dt + 1) * 16 + l) * 32 + half * 16);
      o[dt] = wmma_bf16(pa, bv, o[dt]);
      bv = bn;
    }

    __syncthreads();                              // all waves done reading q/v
    if (sb + 1 < nblk) stage_qv(s0 + 32);         // overlap next DMA with loop overhead
  }

  // Epilogue: divide by softmax denominator, store f32 output [B,T,D].
  float inv[8];
#pragma unroll
  for (int r = 0; r < 8; ++r) inv[r] = 1.0f / lrow[r];
  float* obase = out + ((size_t)b * T_ + t0) * D_;
#pragma unroll
  for (int dt = 0; dt < 16; ++dt) {
#pragma unroll
    for (int r = 0; r < 8; ++r)
      obase[(size_t)(r + 8 * half) * D_ + dt * 16 + l] = o[dt][r] * inv[r];
  }
}

// ---------------------------------------------------------------------------
// Host launcher
// ---------------------------------------------------------------------------
extern "C" void kernel_launch(void* const* d_in, const int* in_sizes, int n_in,
                              void* d_out, int out_size, void* d_ws, size_t ws_size,
                              hipStream_t stream) {
  (void)in_sizes; (void)n_in; (void)out_size; (void)ws_size;
  const float* x  = (const float*)d_in[0];
  const float* Wk = (const float*)d_in[1];
  const float* Wq = (const float*)d_in[2];
  const float* Wv = (const float*)d_in[3];
  float* out = (float*)d_out;

  char* ws = (char*)d_ws;
  size_t off = 0;
  auto carve = [&](size_t bytes) -> void* {
    void* p = ws + off;
    off = (off + bytes + 255) & ~(size_t)255;
    return p;
  };
  unsigned short* xb  = (unsigned short*)carve((size_t)M_ * E_ * 2);  // 64 MB
  unsigned short* wtk = (unsigned short*)carve((size_t)D_ * E_ * 2);
  unsigned short* wtq = (unsigned short*)carve((size_t)D_ * E_ * 2);
  unsigned short* wtv = (unsigned short*)carve((size_t)D_ * E_ * 2);
  unsigned short* kb  = (unsigned short*)carve((size_t)M_ * D_ * 2);
  unsigned short* qb  = (unsigned short*)carve((size_t)M_ * D_ * 2);
  unsigned short* vt  = (unsigned short*)carve((size_t)B_ * D_ * T_ * 2);

  // 1) precision conversion (bf16 inputs, fp32 WMMA accumulate)
  {
    const int n = M_ * E_;
    cvt_f32_bf16<<<dim3(n / 256), dim3(256), 0, stream>>>(x, xb, n);
    const int nw = E_ * D_;
    cvt_w_t<<<dim3(nw / 256), dim3(256), 0, stream>>>(Wk, wtk);
    cvt_w_t<<<dim3(nw / 256), dim3(256), 0, stream>>>(Wq, wtq);
    cvt_w_t<<<dim3(nw / 256), dim3(256), 0, stream>>>(Wv, wtv);
  }

  // 2) K/Q/V projections (V written transposed [B,D,T])
  proj_kernel<<<dim3(M_ / 128, D_ / 64, 3), dim3(256), 0, stream>>>(
      xb, wtk, wtq, wtv, kb, qb, vt);

  // 3) causal flash attention
  fa_kernel<<<dim3(T_ / 128, B_), dim3(256), 0, stream>>>(kb, qb, vt, out);
}
